// Attention_13597866459137
// MI455X (gfx1250) — matrix operand order repaired and checked by reading, not compiled, not run on hardware
//
#include <hip/hip_runtime.h>

// ---- problem constants ----
#define BATCH  4
#define DIM    256
#define HEADS  4
#define DH     32
#define HIDDEN 128
#define NPOS   4096          // 64*64
#define QKV_O  384           // 3*HIDDEN
// q scale (32^-0.5) with log2(e) folded in so softmax can use native base-2 exp
#define QSCALE_LOG2E (0.17677669529663687f * 1.4426950408889634f)

typedef __attribute__((ext_vector_type(8)))  _Float16 v8h;
typedef __attribute__((ext_vector_type(16))) _Float16 v16h;
typedef __attribute__((ext_vector_type(8)))  float    v8f;

// A-matrix fragment (16x32 f16): lane L = row (L&15); lanes<16 hold K {0-7,16-23},
// lanes>=16 hold K {8-15,24-31}. `row` points at 32 contiguous f16 of that row.
__device__ __forceinline__ v16h load_a16(const _Float16* row, int half) {
    v8h lo = *(const v8h*)(row + half * 8);
    v8h hi = *(const v8h*)(row + 16 + half * 8);
    return __builtin_shufflevector(lo, hi, 0,1,2,3,4,5,6,7,8,9,10,11,12,13,14,15);
}

// B-matrix fragment (32x16 f16): lane L = column (L&15); lanes<16 hold K {0-7,16-23},
// lanes>=16 hold K {8-15,24-31}. `col` points at 32 contiguous f16 of that column (K-major).
__device__ __forceinline__ v16h load_b16(const _Float16* col, int half) {
    v8h lo = *(const v8h*)(col + half * 8);
    v8h hi = *(const v8h*)(col + 16 + half * 8);
    return __builtin_shufflevector(lo, hi, 0,1,2,3,4,5,6,7,8,9,10,11,12,13,14,15);
}

__device__ __forceinline__ v8f wmma_f16(v16h a, v16h b, v8f c) {
    return __builtin_amdgcn_wmma_f32_16x16x32_f16(false, a, false, b, (short)0, c, false, false);
}

// ---------------------------------------------------------------------------
// K0: LDS-tiled transpose + f32->f16 convert:  x[b][c][n] -> xT[b][n][c]
// ---------------------------------------------------------------------------
__global__ __launch_bounds__(256) void k_transpose_x(const float* __restrict__ x,
                                                     _Float16* __restrict__ xT) {
    __shared__ float tile[32][33];
    const int n0 = blockIdx.x * 32, c0 = blockIdx.y * 32, b = blockIdx.z;
    const int tx = threadIdx.x, ty = threadIdx.y;
    #pragma unroll
    for (int j = 0; j < 32; j += 8)
        tile[ty + j][tx] = x[((size_t)(b * DIM + c0 + ty + j)) * NPOS + n0 + tx];
    __syncthreads();
    #pragma unroll
    for (int j = 0; j < 32; j += 8)
        xT[((size_t)b * NPOS + n0 + ty + j) * DIM + c0 + tx] = (_Float16)tile[tx][ty + j];
}

// ---------------------------------------------------------------------------
// K1: weight conversion; fold q-scale * log2(e) into first 128 rows of W_qkv
// ---------------------------------------------------------------------------
__global__ __launch_bounds__(256) void k_prep_w(const float* __restrict__ Wqkv,
                                                const float* __restrict__ Wout,
                                                _Float16* __restrict__ Wqh,
                                                _Float16* __restrict__ Woh) {
    int idx = blockIdx.x * 256 + threadIdx.x;
    if (idx < QKV_O * DIM) {
        int o = idx >> 8;  // /DIM
        float s = (o < HIDDEN) ? QSCALE_LOG2E : 1.0f;
        Wqh[idx] = (_Float16)(Wqkv[idx] * s);
    }
    if (idx < DIM * HIDDEN) Woh[idx] = (_Float16)Wout[idx];
}

// ---------------------------------------------------------------------------
// K2: QKV GEMM.  D(16n x 16o) = xT(16n x 32c) * Wqkv^T(32c x 16o), K=256 in 8 steps.
// Writes q,k row-major [b][h][n][32] and v transposed [b][h][32][n] (all f16).
// One wave per 16x16 output tile; 4*24*256 = 24576 waves.
// ---------------------------------------------------------------------------
__global__ __launch_bounds__(256) void k_gemm_qkv(const _Float16* __restrict__ xT,
                                                  const _Float16* __restrict__ Wqh,
                                                  _Float16* __restrict__ qh,
                                                  _Float16* __restrict__ kh,
                                                  _Float16* __restrict__ vT) {
    const int lane = threadIdx.x & 31, half = lane >> 4, lq = lane & 15;
    int w = blockIdx.x * 8 + (threadIdx.x >> 5);
    const int ntile = w % 256; w /= 256;
    const int otile = w % 24;  const int b = w / 24;

    const _Float16* arow = xT + ((size_t)b * NPOS + ntile * 16 + lq) * DIM;
    const _Float16* bcol = Wqh + (size_t)(otile * 16 + lq) * DIM;
    v8f acc = {};
    #pragma unroll
    for (int kc = 0; kc < DIM; kc += 32)
        acc = wmma_f16(load_a16(arow + kc, half), load_b16(bcol + kc, half), acc);

    const int o = otile * 16 + lq;           // output channel for this lane
    #pragma unroll
    for (int i = 0; i < 8; i++) {
        const int n = ntile * 16 + i + 8 * half;
        const _Float16 val = (_Float16)acc[i];
        if (o < HIDDEN) {                    // q
            int hd = o >> 5, d = o & 31;
            qh[(((size_t)(b * HEADS + hd)) * NPOS + n) * DH + d] = val;
        } else if (o < 2 * HIDDEN) {         // k
            int ov = o - HIDDEN, hd = ov >> 5, d = ov & 31;
            kh[(((size_t)(b * HEADS + hd)) * NPOS + n) * DH + d] = val;
        } else {                             // v (transposed: [d][n])
            int ov = o - 2 * HIDDEN, hd = ov >> 5, d = ov & 31;
            vT[(((size_t)(b * HEADS + hd)) * DH + d) * NPOS + n] = val;
        }
    }
}

// ---------------------------------------------------------------------------
// K3: flash attention. One wave owns 16 queries; loops 4096 keys in 64-chunks.
// S^T = K*Q^T (4 WMMAs, scores already in log2 units) -> one online-softmax
// update (per-lane max over 32 regs + one shfl_xor(16), exp2f) -> the score
// accumulators convert in place into the two 16x32 A-fragments for
// O += P*V (4 WMMAs). One rescale broadcast (8 ds_bpermute) per 64 keys.
// ---------------------------------------------------------------------------
__global__ __launch_bounds__(256) void k_flash(const _Float16* __restrict__ qh,
                                               const _Float16* __restrict__ kh,
                                               const _Float16* __restrict__ vT,
                                               _Float16* __restrict__ aoT) {
    const int lane = threadIdx.x & 31, half = lane >> 4, lq = lane & 15;
    int w = blockIdx.x * 8 + (threadIdx.x >> 5);
    const int qtile = w & 255; w >>= 8;
    const int hd = w & 3;      const int b = w >> 2;

    const _Float16* qp = qh + (((size_t)(b * HEADS + hd)) * NPOS + qtile * 16) * DH;
    const _Float16* kp = kh + ((size_t)(b * HEADS + hd)) * NPOS * DH;
    const _Float16* vp = vT + ((size_t)(b * HEADS + hd)) * DH * NPOS;

    // B_Q: Q^T column lq = Q row lq (32 contiguous f16), resident all loop.
    const v16h bq = load_b16(qp + lq * DH, half);

    v8f O0 = {}, O1 = {};
    float m = -3.0e38f, l = 0.0f;

    for (int kb = 0; kb < NPOS; kb += 64) {
        // S^T tiles: rows = keys, cols = queries. Lane -> query lq.
        // s[j][i] = score(key kb + 16*j + 8*half + i, query lq), log2 domain.
        v8f s[4];
        #pragma unroll
        for (int j = 0; j < 4; j++) {
            v16h ak = load_a16(kp + (kb + 16 * j + lq) * DH, half);
            v8f z = {};
            s[j] = wmma_f16(ak, bq, z);
        }

        // online softmax for query lq (lane pair L, L^16), base-2 domain
        float cmax = s[0][0];
        #pragma unroll
        for (int j = 0; j < 4; j++)
            #pragma unroll
            for (int i = 0; i < 8; i++) cmax = fmaxf(cmax, s[j][i]);
        cmax = fmaxf(cmax, __shfl_xor(cmax, 16, 32));
        const float mn = fmaxf(m, cmax);
        const float f  = exp2f(m - mn);

        float psum = 0.0f;
        v16h pa0, pa1;                    // A-fragments of P for keys kb..+31 / +32..+63
        #pragma unroll
        for (int i = 0; i < 8; i++) {
            float p0 = exp2f(s[0][i] - mn); psum += p0; pa0[i]     = (_Float16)p0;
            float p1 = exp2f(s[1][i] - mn); psum += p1; pa0[8 + i] = (_Float16)p1;
            float p2 = exp2f(s[2][i] - mn); psum += p2; pa1[i]     = (_Float16)p2;
            float p3 = exp2f(s[3][i] - mn); psum += p3; pa1[8 + i] = (_Float16)p3;
        }
        psum += __shfl_xor(psum, 16, 32);
        l = l * f + psum;
        m = mn;

        // rescale O rows (row r = i + 8*half carries query r; f lives in lane r)
        #pragma unroll
        for (int i = 0; i < 8; i++) {
            const float fr = __shfl(f, i + 8 * half, 32);
            O0[i] *= fr; O1[i] *= fr;
        }

        // O += P * V  (B from vT: column = dim, contiguous keys)
        v16h bv0a = load_b16(vp + (size_t)lq * NPOS + kb, half);              // dims 0-15,  keys kb..+31
        v16h bv1a = load_b16(vp + (size_t)(lq + 16) * NPOS + kb, half);       // dims 16-31, keys kb..+31
        v16h bv0b = load_b16(vp + (size_t)lq * NPOS + kb + 32, half);         // dims 0-15,  keys +32..+63
        v16h bv1b = load_b16(vp + (size_t)(lq + 16) * NPOS + kb + 32, half);  // dims 16-31, keys +32..+63
        O0 = wmma_f16(pa0, bv0a, O0);
        O1 = wmma_f16(pa0, bv1a, O1);
        O0 = wmma_f16(pa1, bv0b, O0);
        O1 = wmma_f16(pa1, bv1b, O1);
    }

    // epilogue: normalize by l (per output row) and store aoT[b][n][h*32+d]
    #pragma unroll
    for (int i = 0; i < 8; i++) {
        const float inv = 1.0f / __shfl(l, i + 8 * half, 32);
        const int q = qtile * 16 + i + 8 * half;
        const size_t o = ((size_t)b * NPOS + q) * HIDDEN + hd * DH;
        aoT[o + lq]      = (_Float16)(O0[i] * inv);
        aoT[o + 16 + lq] = (_Float16)(O1[i] * inv);
    }
}

// ---------------------------------------------------------------------------
// K4: out-projection GEMM. D(16n x 16o) = aoT(16n x 32c) * Wout^T, K=128 (4 steps).
// Adds b_out and stores pre-BN y[b][o][n] f32. 4*16*256 = 16384 waves.
// ---------------------------------------------------------------------------
__global__ __launch_bounds__(256) void k_gemm_out(const _Float16* __restrict__ aoT,
                                                  const _Float16* __restrict__ Woh,
                                                  const float* __restrict__ bout,
                                                  float* __restrict__ yb) {
    const int lane = threadIdx.x & 31, half = lane >> 4, lq = lane & 15;
    int w = blockIdx.x * 8 + (threadIdx.x >> 5);
    const int ntile = w % 256; w /= 256;
    const int otile = w % 16;  const int b = w / 16;

    const _Float16* arow = aoT + ((size_t)b * NPOS + ntile * 16 + lq) * HIDDEN;
    const _Float16* bcol = Woh + (size_t)(otile * 16 + lq) * HIDDEN;
    v8f acc = {};
    #pragma unroll
    for (int kc = 0; kc < HIDDEN; kc += 32)
        acc = wmma_f16(load_a16(arow + kc, half), load_b16(bcol + kc, half), acc);

    const int o = otile * 16 + lq;
    const float bias = bout[o];
    #pragma unroll
    for (int i = 0; i < 8; i++) {
        const int n = ntile * 16 + i + 8 * half;
        yb[((size_t)(b * DIM + o)) * NPOS + n] = acc[i] + bias;
    }
}

// ---------------------------------------------------------------------------
// K5: BatchNorm stats per channel (mean + 1/sqrt(var+eps) over b,n = 16384)
// ---------------------------------------------------------------------------
__global__ __launch_bounds__(256) void k_bn_stats(const float* __restrict__ yb,
                                                  float* __restrict__ st) {
    const int c = blockIdx.x, t = threadIdx.x;
    float s = 0.0f, ss = 0.0f;
    for (int i = t; i < BATCH * NPOS; i += 256) {
        const int b = i >> 12, n = i & (NPOS - 1);
        const float v = yb[((size_t)(b * DIM + c)) * NPOS + n];
        s += v; ss += v * v;
    }
    __shared__ float shs[256], shq[256];
    shs[t] = s; shq[t] = ss; __syncthreads();
    for (int step = 128; step > 0; step >>= 1) {
        if (t < step) { shs[t] += shs[t + step]; shq[t] += shq[t + step]; }
        __syncthreads();
    }
    if (t == 0) {
        const float inv_n = 1.0f / (float)(BATCH * NPOS);
        const float mean = shs[0] * inv_n;
        const float var  = shq[0] * inv_n - mean * mean;
        st[c]       = mean;
        st[DIM + c] = rsqrtf(var + 1e-5f);
    }
}

// ---------------------------------------------------------------------------
// K6: BN apply -> d_out (f32, layout b,256,64,64 == b,256,4096)
// ---------------------------------------------------------------------------
__global__ __launch_bounds__(256) void k_bn_apply(const float* __restrict__ yb,
                                                  const float* __restrict__ st,
                                                  const float* __restrict__ gamma,
                                                  const float* __restrict__ beta,
                                                  float* __restrict__ out) {
    const size_t idx = (size_t)blockIdx.x * 256 + threadIdx.x;
    if (idx >= (size_t)BATCH * DIM * NPOS) return;
    const int c = (int)((idx >> 12) & (DIM - 1));
    out[idx] = (yb[idx] - st[c]) * st[DIM + c] * gamma[c] + beta[c];
}

// ---------------------------------------------------------------------------
extern "C" void kernel_launch(void* const* d_in, const int* in_sizes, int n_in,
                              void* d_out, int out_size, void* d_ws, size_t ws_size,
                              hipStream_t stream) {
    const float* x     = (const float*)d_in[0];
    const float* Wqkv  = (const float*)d_in[1];
    const float* Wout  = (const float*)d_in[2];
    const float* bout  = (const float*)d_in[3];
    const float* gamma = (const float*)d_in[4];
    const float* beta  = (const float*)d_in[5];
    float* out = (float*)d_out;

    char* ws = (char*)d_ws;
    _Float16* xT  = (_Float16*)(ws + 0);          //  8388608 B
    _Float16* Wqh = (_Float16*)(ws + 8388608);    //   196608 B
    _Float16* Woh = (_Float16*)(ws + 8585216);    //    65536 B
    _Float16* qh  = (_Float16*)(ws + 8650752);    //  4194304 B
    _Float16* kh  = (_Float16*)(ws + 12845056);   //  4194304 B
    _Float16* vT  = (_Float16*)(ws + 17039360);   //  4194304 B
    _Float16* aoT = (_Float16*)(ws + 21233664);   //  4194304 B
    float*    yb  = (float*)   (ws + 25427968);   // 16777216 B
    float*    st  = (float*)   (ws + 42205184);   //     2048 B

    k_transpose_x<<<dim3(NPOS / 32, DIM / 32, BATCH), dim3(32, 8), 0, stream>>>(x, xT);
    k_prep_w<<<(QKV_O * DIM + 255) / 256, 256, 0, stream>>>(Wqkv, Wout, Wqh, Woh);
    k_gemm_qkv<<<(BATCH * 24 * 256) / 8, 256, 0, stream>>>(xT, Wqh, qh, kh, vT);
    k_flash<<<(BATCH * HEADS * 256) / 8, 256, 0, stream>>>(qh, kh, vT, aoT);
    k_gemm_out<<<(BATCH * 16 * 256) / 8, 256, 0, stream>>>(aoT, Woh, bout, yb);
    k_bn_stats<<<DIM, 256, 0, stream>>>(yb, st);
    k_bn_apply<<<(BATCH * DIM * NPOS + 255) / 256, 256, 0, stream>>>(yb, st, gamma, beta, out);
}
